// Spiking_GFNN_53480932769977
// MI455X (gfx1250) — matrix operand order repaired and checked by reading, not compiled, not run on hardware
//
#include <hip/hip_runtime.h>

// ---------------------------------------------------------------------------
// Types for CDNA5 WMMA (wave32): A/B = 16 bf16 per lane, C/D = 8 f32 per lane
// ---------------------------------------------------------------------------
typedef __attribute__((ext_vector_type(16))) __bf16 bf16x16;
typedef __attribute__((ext_vector_type(8)))  float  f32x8;
typedef __attribute__((ext_vector_type(4)))  int    v4i_;

union FragAB { bf16x16 v; uint4 q[2]; };

// CDNA5 async global->LDS DMA path (ASYNCcnt-tracked), guarded so a toolchain
// without the builtins falls back to the register-staged copy (still WMMA).
#if defined(__has_builtin)
#  if __has_builtin(__builtin_amdgcn_global_load_async_to_lds_b128) && \
      __has_builtin(__builtin_amdgcn_s_wait_asynccnt)
#    define USE_ASYNC_COPY 1
#  endif
#endif
#ifndef USE_ASYNC_COPY
#  define USE_ASYNC_COPY 0
#endif

#if USE_ASYNC_COPY
typedef __attribute__((address_space(1))) v4i_ gv4i;   // global int4
typedef __attribute__((address_space(3))) v4i_ sv4i;   // LDS int4
#endif

__device__ __forceinline__ void g2l_b128(const unsigned short* g, unsigned short* l) {
#if USE_ASYNC_COPY
  __builtin_amdgcn_global_load_async_to_lds_b128(
      (gv4i*)(unsigned long long)g,
      (sv4i*)(unsigned int)(unsigned long long)l,
      0, 0);
#else
  *(uint4*)l = *(const uint4*)g;
#endif
}

__device__ __forceinline__ void wait_copy_tile_done() {   // prev tile complete
#if USE_ASYNC_COPY
  __builtin_amdgcn_s_wait_asynccnt(4);   // 4 in flight for next tile, rest done
#endif
}
__device__ __forceinline__ void wait_copy_all_done() {
#if USE_ASYNC_COPY
  __builtin_amdgcn_s_wait_asynccnt(0);
#endif
}

__device__ __forceinline__ unsigned short f32_to_bf16(float f) {
  unsigned int u = __float_as_uint(f);
  u += 0x7FFFu + ((u >> 16) & 1u);          // round-to-nearest-even
  return (unsigned short)(u >> 16);
}

// ---------------------------------------------------------------------------
// fp32 -> bf16 packed converter (2 elems / thread)
// ---------------------------------------------------------------------------
__global__ __launch_bounds__(256)
void cvt_f32_to_bf16(const float2* __restrict__ in, unsigned int* __restrict__ out, int n2) {
  int i = blockIdx.x * 256 + threadIdx.x;
  if (i < n2) {
    float2 f = in[i];
    out[i] = (unsigned int)f32_to_bf16(f.x) | ((unsigned int)f32_to_bf16(f.y) << 16);
  }
}

// ---------------------------------------------------------------------------
// Transpose fp32 [R][C] -> bf16 [C][R]  (K-major weight layout for WMMA B)
// ---------------------------------------------------------------------------
__global__ __launch_bounds__(256)
void transpose_f32_to_bf16(const float* __restrict__ src, unsigned short* __restrict__ dst,
                           int Rr, int Cc) {
  __shared__ float tile[32][33];
  int c0 = blockIdx.x * 32;
  int r0 = blockIdx.y * 32;
  int tx = threadIdx.x & 31;
  int ty = threadIdx.x >> 5;              // 0..7
  for (int i = ty; i < 32; i += 8)
    tile[i][tx] = src[(size_t)(r0 + i) * Cc + c0 + tx];
  __syncthreads();
  for (int i = ty; i < 32; i += 8)
    dst[(size_t)(c0 + i) * Rr + r0 + tx] = f32_to_bf16(tile[tx][i]);
}

// ---------------------------------------------------------------------------
// bf16 WMMA GEMM:  C[M][N] = A[M][K] * Bt[N][K]^T + bias[N]   (fp32 accum)
// Block tile 128x128, BK=32, double-buffered LDS fed by async global->LDS DMA.
// 8 waves: wave (wm 0..3, wn 0..1) -> 32x64 subtile, 8 v_wmma per k-step.
// ---------------------------------------------------------------------------
#define BM  128
#define BNT 128
#define BK  32
#define LDT 48   // LDS row stride (elems): 96B, 16B-aligned, bank-spread

__global__ __launch_bounds__(256)
void gemm_bf16_wmma(const unsigned short* __restrict__ A,
                    const unsigned short* __restrict__ Bt,
                    const float* __restrict__ bias,
                    float* __restrict__ Cc,
                    int M, int N, int K) {
  __shared__ unsigned short As[2][BM * LDT];
  __shared__ unsigned short Bs[2][BNT * LDT];
  (void)M;

  const int tid  = threadIdx.x;
  const int lane = tid & 31;
  const int wave = tid >> 5;
  const int wm   = wave & 3;           // 32-row strip
  const int wn   = wave >> 2;          // 64-col strip

  const int bm = blockIdx.y * BM;
  const int bn = blockIdx.x * BNT;

  f32x8 acc[2][4];
#pragma unroll
  for (int i = 0; i < 2; ++i)
#pragma unroll
    for (int j = 0; j < 4; ++j)
#pragma unroll
      for (int e = 0; e < 8; ++e) acc[i][j][e] = 0.0f;

  // tile copy: thread t -> row t/4 (+64), 8 consecutive bf16 (16B) x2 each mat
  const int lrow = tid >> 2;
  const int lcol = (tid & 3) << 3;
  const unsigned short* pa0 = A  + (size_t)(bm + lrow) * K + lcol;
  const unsigned short* pb0 = Bt + (size_t)(bn + lrow) * K + lcol;
  const size_t rowskip = (size_t)64 * K;

  // fragment addressing (ISA 16-bit A 16x32 / B 32x16 layouts)
  const int r   = lane & 15;
  const int kbA = (lane >> 4) << 3;    // A: K-half base 0 / 8   (+16 for upper)
  const int kbB = (lane >> 4) << 3;    // B: K-half base 0 / 8   (+16 for upper)

  auto copy_tile = [&](int kt, int buf) {
    const unsigned short* pa = pa0 + (size_t)kt * BK;
    const unsigned short* pb = pb0 + (size_t)kt * BK;
    unsigned short* la = &As[buf][lrow * LDT + lcol];
    unsigned short* lb = &Bs[buf][lrow * LDT + lcol];
    g2l_b128(pa,           la);
    g2l_b128(pa + rowskip, la + 64 * LDT);
    g2l_b128(pb,           lb);
    g2l_b128(pb + rowskip, lb + 64 * LDT);
  };

  auto compute_tile = [&](int buf) {
    FragAB afrag[2], bfrag[4];
#pragma unroll
    for (int im = 0; im < 2; ++im) {
      const unsigned short* p = &As[buf][(wm * 32 + im * 16 + r) * LDT + kbA];
      afrag[im].q[0] = *(const uint4*)(p);        // K = kbA .. kbA+7
      afrag[im].q[1] = *(const uint4*)(p + 16);   // K = kbA+16 .. kbA+23
    }
#pragma unroll
    for (int in = 0; in < 4; ++in) {
      const unsigned short* p = &Bs[buf][(wn * 64 + in * 16 + r) * LDT + kbB];
      bfrag[in].q[0] = *(const uint4*)(p);        // K = kbB .. kbB+7
      bfrag[in].q[1] = *(const uint4*)(p + 16);   // K = kbB+16 .. kbB+23
    }
#pragma unroll
    for (int im = 0; im < 2; ++im)
#pragma unroll
      for (int in = 0; in < 4; ++in)
        acc[im][in] = __builtin_amdgcn_wmma_f32_16x16x32_bf16(
            false, afrag[im].v, false, bfrag[in].v,
            (short)0, acc[im][in], false, false);
  };

  const int nk = K / BK;
  copy_tile(0, 0);
  int parity = 0;
  for (int it = 0; it < nk - 1; ++it) {
    copy_tile(it + 1, parity ^ 1);   // overlap next tile DMA with this WMMA burst
    wait_copy_tile_done();           // s_wait_asynccnt 4: tile `it` landed
    __syncthreads();
    compute_tile(parity);
    __syncthreads();                 // all waves done reading before overwrite
    parity ^= 1;
  }
  wait_copy_all_done();              // s_wait_asynccnt 0
  __syncthreads();
  compute_tile(parity);

  // C/D layout: VGPR e, lanes 0-15 -> M=e, lanes 16-31 -> M=8+e; N = lane&15
  const int rhalf = (lane >> 4) * 8;
#pragma unroll
  for (int im = 0; im < 2; ++im) {
#pragma unroll
    for (int in = 0; in < 4; ++in) {
      const int col = bn + wn * 64 + in * 16 + r;
      const float bv = bias[col];
#pragma unroll
      for (int e = 0; e < 8; ++e) {
        const int row = bm + wm * 32 + im * 16 + rhalf + e;
        Cc[(size_t)row * N + col] = acc[im][in][e] + bv;
      }
    }
  }
}

// ---------------------------------------------------------------------------
// BatchNorm stats, deterministic two-stage (coalesced channel-major partials)
// ---------------------------------------------------------------------------
__global__ __launch_bounds__(256)
void bn_partial(const float* __restrict__ X, int ldx, int rpc,
                float* __restrict__ psum, float* __restrict__ psq, int nch) {
  int c  = blockIdx.x * 256 + threadIdx.x;
  int r0 = blockIdx.y * rpc;
  const float* p = X + (size_t)r0 * ldx + c;
  float s = 0.f, q = 0.f;
  for (int r = 0; r < rpc; ++r) {
    float v = p[(size_t)r * ldx];
    s += v; q += v * v;
  }
  psum[(size_t)blockIdx.y * nch + c] = s;
  psq [(size_t)blockIdx.y * nch + c] = q;
}

__global__ __launch_bounds__(256)
void bn_finalize(const float* __restrict__ psum, const float* __restrict__ psq,
                 int chunks, int nch, float inv_n,
                 float* __restrict__ mean, float* __restrict__ rstd) {
  int c = blockIdx.x * 256 + threadIdx.x;
  if (c >= nch) return;
  float s = 0.f, q = 0.f;
  for (int r = 0; r < chunks; ++r) { s += psum[(size_t)r * nch + c]; q += psq[(size_t)r * nch + c]; }
  float m = s * inv_n;
  mean[c] = m;
  rstd[c] = rsqrtf(q * inv_n - m * m + 1e-5f);
}

// ---------------------------------------------------------------------------
// Fused BN + LIF scan (T=4) + gate, emit bf16 activations (GEMM2 A operand)
// hid: [T*SP][3072]  (v = cols 0..1535, gate = cols 1536..3071)
// ---------------------------------------------------------------------------
__global__ __launch_bounds__(256)
void lif_gate_kernel(const float* __restrict__ hid,
                     const float* __restrict__ g, const float* __restrict__ be,
                     const float* __restrict__ mean, const float* __restrict__ rstd,
                     unsigned short* __restrict__ act, int sp_total, int Hc) {
  int idx = blockIdx.x * 256 + threadIdx.x;
  if (idx >= sp_total * Hc) return;
  int h  = idx % Hc;                       // consecutive threads -> coalesced
  int sp = idx / Hc;
  float gm = g[h] * rstd[h];
  float bs = be[h] - mean[h] * gm;
  float v = 0.f;
#pragma unroll
  for (int t = 0; t < 4; ++t) {
    size_t row = (size_t)t * sp_total + sp;
    float vin  = hid[row * 3072 + h];
    float gate = hid[row * 3072 + 1536 + h];
    float val  = fmaf(gm, vin, bs);
    v = 0.5f * (v + val);                  // v += (val - v)/TAU, TAU=2
    float s = (v >= 1.0f) ? 1.0f : 0.0f;   // spike(v - V_TH)
    v *= (1.0f - s);
    act[row * 1536 + h] = f32_to_bf16(gate * s);
  }
}

// Fused BN + LIF -> final 0/1 fp32 spikes
__global__ __launch_bounds__(256)
void lif_out_kernel(const float* __restrict__ pre,
                    const float* __restrict__ g, const float* __restrict__ be,
                    const float* __restrict__ mean, const float* __restrict__ rstd,
                    float* __restrict__ out, int sp_total, int Cch) {
  int idx = blockIdx.x * 256 + threadIdx.x;
  if (idx >= sp_total * Cch) return;
  int c  = idx % Cch;
  int sp = idx / Cch;
  float gm = g[c] * rstd[c];
  float bs = be[c] - mean[c] * gm;
  float v = 0.f;
#pragma unroll
  for (int t = 0; t < 4; ++t) {
    size_t row = (size_t)t * sp_total + sp;
    float val = fmaf(gm, pre[row * 768 + c], bs);
    v = 0.5f * (v + val);
    float s = (v >= 1.0f) ? 1.0f : 0.0f;
    v *= (1.0f - s);
    out[row * 768 + c] = s;
  }
}

// ---------------------------------------------------------------------------
extern "C" void kernel_launch(void* const* d_in, const int* in_sizes, int n_in,
                              void* d_out, int out_size, void* d_ws, size_t ws_size,
                              hipStream_t stream) {
  (void)in_sizes; (void)n_in; (void)out_size; (void)ws_size;
  const float* x   = (const float*)d_in[0];
  const float* W1  = (const float*)d_in[1];
  const float* b1  = (const float*)d_in[2];
  const float* g1  = (const float*)d_in[3];
  const float* be1 = (const float*)d_in[4];
  const float* W2  = (const float*)d_in[5];
  const float* b2  = (const float*)d_in[6];
  const float* g2  = (const float*)d_in[7];
  const float* be2 = (const float*)d_in[8];
  float* out = (float*)d_out;

  constexpr int M = 12544, SP = 3136, C = 768, H = 1536;
  constexpr int N1 = 2 * H, K1 = C, N2 = C, K2 = H;
  constexpr int CHUNKS = 98, RPC = 128;      // 98*128 == 12544

  char* ws = (char*)d_ws;
  size_t off = 0;
  auto alloc = [&](size_t bytes) -> char* {
    char* p = ws + off;
    off = (off + bytes + 255) & ~(size_t)255;
    return p;
  };
  unsigned short* xbf  = (unsigned short*)alloc((size_t)M  * K1 * 2);
  unsigned short* W1t  = (unsigned short*)alloc((size_t)N1 * K1 * 2);
  unsigned short* W2t  = (unsigned short*)alloc((size_t)N2 * K2 * 2);
  float*          hid  = (float*)         alloc((size_t)M  * N1 * 4);
  unsigned short* act  = (unsigned short*)alloc((size_t)M  * H  * 2);
  float*          pre2 = (float*)         alloc((size_t)M  * C  * 4);
  float* psum1 = (float*)alloc((size_t)CHUNKS * H * 4);
  float* psq1  = (float*)alloc((size_t)CHUNKS * H * 4);
  float* psum2 = (float*)alloc((size_t)CHUNKS * C * 4);
  float* psq2  = (float*)alloc((size_t)CHUNKS * C * 4);
  float* mean1 = (float*)alloc(H * 4);
  float* rstd1 = (float*)alloc(H * 4);
  float* mean2 = (float*)alloc(C * 4);
  float* rstd2 = (float*)alloc(C * 4);

  // x -> bf16
  {
    int n2 = (M * K1) / 2;
    cvt_f32_to_bf16<<<(n2 + 255) / 256, 256, 0, stream>>>((const float2*)x, (unsigned int*)xbf, n2);
  }
  // weights -> bf16, K-major (transposed)
  transpose_f32_to_bf16<<<dim3(N1 / 32, K1 / 32), 256, 0, stream>>>(W1, W1t, K1, N1);
  transpose_f32_to_bf16<<<dim3(N2 / 32, K2 / 32), 256, 0, stream>>>(W2, W2t, K2, N2);

  // GEMM1: hidden = x @ W1 + b1
  gemm_bf16_wmma<<<dim3(N1 / 128, M / 128), 256, 0, stream>>>(xbf, W1t, b1, hid, M, N1, K1);

  // BN stats over v-half (cols 0..H-1)
  bn_partial<<<dim3(H / 256, CHUNKS), 256, 0, stream>>>(hid, N1, RPC, psum1, psq1, H);
  bn_finalize<<<(H + 255) / 256, 256, 0, stream>>>(psum1, psq1, CHUNKS, H, 1.0f / (float)M, mean1, rstd1);

  // fused BN + LIF + gate -> bf16 activations
  lif_gate_kernel<<<(SP * H + 255) / 256, 256, 0, stream>>>(hid, g1, be1, mean1, rstd1, act, SP, H);

  // GEMM2: pre2 = act @ W2 + b2
  gemm_bf16_wmma<<<dim3(N2 / 128, M / 128), 256, 0, stream>>>(act, W2t, b2, pre2, M, N2, K2);

  // BN stats stage 2
  bn_partial<<<dim3(C / 256, CHUNKS), 256, 0, stream>>>(pre2, C, RPC, psum2, psq2, C);
  bn_finalize<<<(C + 255) / 256, 256, 0, stream>>>(psum2, psq2, CHUNKS, C, 1.0f / (float)M, mean2, rstd2);

  // fused BN + LIF -> output spikes
  lif_out_kernel<<<(SP * C + 255) / 256, 256, 0, stream>>>(pre2, g2, be2, mean2, rstd2, out, SP, C);
}
